// SelfAttention_72138270703785
// MI455X (gfx1250) — matrix operand order repaired and checked by reading, not compiled, not run on hardware
//
#include <hip/hip_runtime.h>

// ---------------------------------------------------------------------------
// MI455X (gfx1250) self-attention: QKV GEMM -> RoPE -> flash attention -> proj
// v_wmma_f32_16x16x32_f16 everywhere; async global->LDS staging; DPP reductions
// ---------------------------------------------------------------------------

typedef _Float16 v16h __attribute__((ext_vector_type(16)));
typedef _Float16 v8h  __attribute__((ext_vector_type(8)));
typedef _Float16 v2h  __attribute__((ext_vector_type(2)));
typedef float    v8f  __attribute__((ext_vector_type(8)));

#define WMMA_F16(A_, B_, C_) \
  __builtin_amdgcn_wmma_f32_16x16x32_f16(false, (A_), false, (B_), (short)0, (C_), false, false)

static __device__ __forceinline__ v16h cat16(v8h lo, v8h hi) {
  v16h r;
#pragma unroll
  for (int j = 0; j < 8; ++j) { r[j] = lo[j]; r[j + 8] = hi[j]; }
  return r;
}

// ---- async global->LDS (CDNA5): per-lane 16B copy, tracked by ASYNCcnt -----
static __device__ __forceinline__ void async_b128(const void* gaddr, void* lds) {
  unsigned lo = (unsigned)(unsigned long long)lds;  // flat LDS addr[31:0] = LDS offset
  asm volatile("global_load_async_to_lds_b128 %0, %1, off"
               :: "v"(lo), "v"(gaddr) : "memory");
}
static __device__ __forceinline__ void wait_async0() {
  asm volatile("s_wait_asynccnt 0x0" ::: "memory");
}

// ---- DPP reductions over a 16-lane group (cols of one C/D row) -------------
// xor1/2/4 via DPP8 (intra-8-lane), xor8 via DPP16 row_ror:8 (i^8 in 16-row)
#define DPP8_XOR1 0xDE54C1
#define DPP8_XOR2 0xB3E21A
#define DPP8_XOR4 0x688FAC
static __device__ __forceinline__ float red_max16(float x) {
  int t;
  t = __builtin_amdgcn_mov_dpp8(__float_as_int(x), DPP8_XOR1); x = fmaxf(x, __int_as_float(t));
  t = __builtin_amdgcn_mov_dpp8(__float_as_int(x), DPP8_XOR2); x = fmaxf(x, __int_as_float(t));
  t = __builtin_amdgcn_mov_dpp8(__float_as_int(x), DPP8_XOR4); x = fmaxf(x, __int_as_float(t));
  t = __builtin_amdgcn_update_dpp(__float_as_int(x), __float_as_int(x),
                                  0x128 /*row_ror:8*/, 0xF, 0xF, true);
  x = fmaxf(x, __int_as_float(t));
  return x;
}
static __device__ __forceinline__ float red_sum16(float x) {
  int t;
  t = __builtin_amdgcn_mov_dpp8(__float_as_int(x), DPP8_XOR1); x += __int_as_float(t);
  t = __builtin_amdgcn_mov_dpp8(__float_as_int(x), DPP8_XOR2); x += __int_as_float(t);
  t = __builtin_amdgcn_mov_dpp8(__float_as_int(x), DPP8_XOR4); x += __int_as_float(t);
  t = __builtin_amdgcn_update_dpp(__float_as_int(x), __float_as_int(x),
                                  0x128 /*row_ror:8*/, 0xF, 0xF, true);
  x += __int_as_float(t);
  return x;
}

// Problem constants: B=4, T=2048, C=1024, H=16, hd=64
#define BB 4
#define TT 2048
#define CC 1024
#define NH 16
#define HD 64

// ---------------------------------------------------------------------------
// 1) elementwise f32 -> f16 convert
// ---------------------------------------------------------------------------
__global__ void cvt_f32_to_f16(const float* __restrict__ in, _Float16* __restrict__ out, int n) {
  int i = blockIdx.x * blockDim.x + threadIdx.x;
  int stride = gridDim.x * blockDim.x;
  for (; i < n; i += stride) out[i] = (_Float16)in[i];
}

// ---------------------------------------------------------------------------
// 2) f16 GEMM: C(MxN) = A(MxK f16 rm) * B(KxN f16 rm); 64x64 block, 4 waves
// ---------------------------------------------------------------------------
template <typename OUT>
__global__ void __launch_bounds__(128)
gemm_f16_kernel(const _Float16* __restrict__ A, const _Float16* __restrict__ Bm,
                OUT* __restrict__ Cm, int M, int N, int K) {
  __shared__ _Float16 Al[64][40];   // A tile [m][k], padded rows (16B aligned)
  __shared__ _Float16 Bl[64][40];   // B tile transposed [n][k]

  const int t    = threadIdx.x;
  const int w    = t >> 5;
  const int lane = t & 31;
  const int nn   = lane & 15;
  const int hi   = (lane >> 4) & 1;
  const int m0   = blockIdx.y * 64;
  const int n0   = blockIdx.x * 64;

  v8f zero = {};
  v8f acc[4];
#pragma unroll
  for (int c = 0; c < 4; ++c) acc[c] = zero;

  for (int kk = 0; kk < K; kk += 32) {
    __syncthreads();  // previous iteration's fragment reads done
    {   // A tile 64x32: contiguous per-thread 32B -> async global->LDS
      int r = t >> 1, c16 = (t & 1) * 16;
      const _Float16* g = A + (size_t)(m0 + r) * K + kk + c16;
      async_b128(g,     &Al[r][c16]);
      async_b128(g + 8, &Al[r][c16 + 8]);
    }
    {   // B tile 32x64 -> transposed [n][k]; 2 k-rows per thread, b32 stores
      int kp = t >> 3;            // 0..15 -> k rows 2kp, 2kp+1
      int n8 = (t & 7) * 8;       // 8 n per thread
      const v8h* g0 = (const v8h*)(Bm + (size_t)(kk + 2 * kp) * N + n0 + n8);
      const v8h* g1 = (const v8h*)(Bm + (size_t)(kk + 2 * kp + 1) * N + n0 + n8);
      v8h b0 = *g0, b1 = *g1;
#pragma unroll
      for (int j = 0; j < 8; ++j) {
        v2h p; p[0] = b0[j]; p[1] = b1[j];
        *(v2h*)&Bl[n8 + j][2 * kp] = p;
      }
    }
    wait_async0();
    __syncthreads();

    // A frag: M = nn; K = 16*hi + j, contiguous in Al[m][*]
    v16h af = cat16(*(const v8h*)&Al[16 * w + nn][16 * hi],
                    *(const v8h*)&Al[16 * w + nn][16 * hi + 8]);
#pragma unroll
    for (int c = 0; c < 4; ++c) {
      // B frag: N = nn; K = 16*hi + j, contiguous in Bl[n][*]
      v16h bf = cat16(*(const v8h*)&Bl[16 * c + nn][16 * hi],
                      *(const v8h*)&Bl[16 * c + nn][16 * hi + 8]);
      acc[c] = WMMA_F16(af, bf, acc[c]);
    }
  }

  // C/D layout: element r -> row r + 8*hi, col nn
#pragma unroll
  for (int r = 0; r < 8; ++r) {
    int row = m0 + 16 * w + r + 8 * hi;
#pragma unroll
    for (int c = 0; c < 4; ++c) {
      int col = n0 + 16 * c + nn;
      Cm[(size_t)row * N + col] = (OUT)acc[c][r];
    }
  }
}

// ---------------------------------------------------------------------------
// 3) RoPE + head split/transpose; folds 1/sqrt(hd)=0.125 into Q
// ---------------------------------------------------------------------------
__global__ void rope_kernel(const _Float16* __restrict__ qkv,
                            _Float16* __restrict__ Qh,
                            _Float16* __restrict__ Kh,
                            _Float16* __restrict__ Vh) {
  int idx = blockIdx.x * blockDim.x + threadIdx.x;
  if (idx >= BB * TT * NH * (HD / 2)) return;
  int i  = idx & 31;
  int h  = (idx >> 5) & 15;
  int tp = (idx >> 9) & 2047;
  int b  = idx >> 20;

  const _Float16* p = qkv + ((size_t)b * TT + tp) * (3 * CC) + h * HD + 2 * i;
  float q0 = (float)p[0],      q1 = (float)p[1];
  float k0 = (float)p[CC],     k1 = (float)p[CC + 1];
  float v0 = (float)p[2 * CC], v1 = (float)p[2 * CC + 1];

  float inv = __expf(-(2.0f * (float)i / (float)HD) * 9.210340371976184f); // ln(1e4)
  float fr  = (float)tp * inv;
  float cs = __cosf(fr), sn = __sinf(fr);

  size_t o = (((size_t)b * NH + h) * TT + tp) * HD + 2 * i;
  Qh[o]     = (_Float16)(0.125f * (q0 * cs - q1 * sn));
  Qh[o + 1] = (_Float16)(0.125f * (q1 * cs + q0 * sn));
  Kh[o]     = (_Float16)(k0 * cs - k1 * sn);
  Kh[o + 1] = (_Float16)(k1 * cs + k0 * sn);
  Vh[o]     = (_Float16)v0;
  Vh[o + 1] = (_Float16)v1;
}

// ---------------------------------------------------------------------------
// 4) Flash attention: block = 64 q-rows of one (b,h); 4 waves x 16 rows
// ---------------------------------------------------------------------------
__global__ void __launch_bounds__(128)
attn_kernel(const _Float16* __restrict__ Qh, const _Float16* __restrict__ Kh,
            const _Float16* __restrict__ Vh, _Float16* __restrict__ Yh) {
  __shared__ _Float16 Kl[32][72];      // K tile row-major [key][d]
  __shared__ _Float16 Vl[64][40];      // V tile transposed [d][key]
  __shared__ _Float16 Pl[4][16][40];   // per-wave P staging [qrow][key]

  const int t    = threadIdx.x;
  const int w    = t >> 5;
  const int lane = t & 31;
  const int nn   = lane & 15;
  const int hi   = (lane >> 4) & 1;
  const int b    = blockIdx.z, h = blockIdx.y;
  const size_t bh = ((size_t)b * NH + h) * TT;
  const int q0 = blockIdx.x * 64;
  const int qw = q0 + 16 * w;

  // Q fragments held in registers for the whole loop (A-layout, M = nn)
  const _Float16* qrow = Qh + (bh + qw + nn) * HD;
  v16h qf[2];
#pragma unroll
  for (int dc = 0; dc < 2; ++dc)
    qf[dc] = cat16(*(const v8h*)(qrow + 32 * dc + 16 * hi),
                   *(const v8h*)(qrow + 32 * dc + 16 * hi + 8));

  v8f zero = {};
  v8f o[4];
#pragma unroll
  for (int c = 0; c < 4; ++c) o[c] = zero;
  float mrow[8], lrow[8];
#pragma unroll
  for (int r = 0; r < 8; ++r) { mrow[r] = -3.0e38f; lrow[r] = 0.0f; }

  const int kend = q0 + 64;  // causal
  for (int kk = 0; kk < kend; kk += 32) {
    __syncthreads();
    {   // K tile 32x64 row-major: contiguous 32B/thread -> async global->LDS
      int kr = t >> 2, d0 = (t & 3) * 16;
      const _Float16* gk = Kh + (bh + kk + kr) * HD + d0;
      async_b128(gk,     &Kl[kr][d0]);
      async_b128(gk + 8, &Kl[kr][d0 + 8]);
    }
    {   // V tile -> transposed [d][key]; 2 keys per thread, packed b32 stores
      int kp = t >> 3;            // 0..15 -> keys 2kp, 2kp+1
      int d0 = (t & 7) * 8;       // 8 d per thread
      v8h a = *(const v8h*)(Vh + (bh + kk + 2 * kp) * HD + d0);
      v8h c = *(const v8h*)(Vh + (bh + kk + 2 * kp + 1) * HD + d0);
#pragma unroll
      for (int j = 0; j < 8; ++j) {
        v2h p; p[0] = a[j]; p[1] = c[j];
        *(v2h*)&Vl[d0 + j][2 * kp] = p;
      }
    }
    wait_async0();
    __syncthreads();

    // S tiles: s0 = keys [kk,kk+16), s1 = [kk+16,kk+32); K-dim = hd in 2 steps
    v8f s0 = zero, s1 = zero;
#pragma unroll
    for (int dc = 0; dc < 2; ++dc) {
      v16h b0 = cat16(*(const v8h*)&Kl[nn][32 * dc + 16 * hi],
                      *(const v8h*)&Kl[nn][32 * dc + 16 * hi + 8]);
      s0 = WMMA_F16(qf[dc], b0, s0);
      v16h b1 = cat16(*(const v8h*)&Kl[16 + nn][32 * dc + 16 * hi],
                      *(const v8h*)&Kl[16 + nn][32 * dc + 16 * hi + 8]);
      s1 = WMMA_F16(qf[dc], b1, s1);
    }

    // causal mask + online softmax (row = r+8*hi, cols across 16-lane group)
#pragma unroll
    for (int r = 0; r < 8; ++r) {
      int q = qw + r + 8 * hi;
      float a0 = (kk + nn      <= q) ? s0[r] : -3.0e38f;
      float a1 = (kk + 16 + nn <= q) ? s1[r] : -3.0e38f;
      float mx    = red_max16(fmaxf(a0, a1));
      float mnew  = fmaxf(mrow[r], mx);
      float alpha = __expf(mrow[r] - mnew);
      float p0 = __expf(a0 - mnew);
      float p1 = __expf(a1 - mnew);
      float sum = red_sum16(p0 + p1);
      lrow[r] = lrow[r] * alpha + sum;
      mrow[r] = mnew;
#pragma unroll
      for (int c = 0; c < 4; ++c) o[c][r] *= alpha;
      Pl[w][r + 8 * hi][nn]      = (_Float16)p0;
      Pl[w][r + 8 * hi][16 + nn] = (_Float16)p1;
    }
    __syncthreads();  // C-layout -> A-layout conversion of P via LDS

    // O += P(16x32) * V(32x64)
    v16h pf = cat16(*(const v8h*)&Pl[w][nn][16 * hi],
                    *(const v8h*)&Pl[w][nn][16 * hi + 8]);
#pragma unroll
    for (int c = 0; c < 4; ++c) {
      v16h vf = cat16(*(const v8h*)&Vl[16 * c + nn][16 * hi],
                      *(const v8h*)&Vl[16 * c + nn][16 * hi + 8]);
      o[c] = WMMA_F16(pf, vf, o[c]);
    }
  }

  // normalize, write Y in (B,T,C) layout
#pragma unroll
  for (int r = 0; r < 8; ++r) {
    float invl = 1.0f / lrow[r];
    int q = qw + r + 8 * hi;
#pragma unroll
    for (int c = 0; c < 4; ++c) {
      size_t oi = ((size_t)b * TT + q) * CC + (size_t)h * HD + 16 * c + nn;
      Yh[oi] = (_Float16)(o[c][r] * invl);
    }
  }
}

// ---------------------------------------------------------------------------
// launch
// ---------------------------------------------------------------------------
extern "C" void kernel_launch(void* const* d_in, const int* in_sizes, int n_in,
                              void* d_out, int out_size, void* d_ws, size_t ws_size,
                              hipStream_t stream) {
  (void)in_sizes; (void)n_in; (void)out_size; (void)ws_size;
  const float* x      = (const float*)d_in[0];   // (B,T,C)
  const float* W_attn = (const float*)d_in[1];   // (C,3C)
  const float* W_proj = (const float*)d_in[2];   // (C,C)
  float* out = (float*)d_out;                    // (B,T,C)

  const int M   = BB * TT;         // 8192
  const int nX  = M * CC;          // 8,388,608
  const int nWa = CC * 3 * CC;     // 3,145,728
  const int nWp = CC * CC;         // 1,048,576

  char* ws = (char*)d_ws;
  size_t off = 0;
  _Float16* Xh   = (_Float16*)(ws + off); off += (size_t)nX * 2;          // 16 MiB
  _Float16* Wah  = (_Float16*)(ws + off); off += (size_t)nWa * 2;         //  6 MiB
  _Float16* Wph  = (_Float16*)(ws + off); off += (size_t)nWp * 2;         //  2 MiB
  _Float16* QKVh = (_Float16*)(ws + off); off += (size_t)M * 3 * CC * 2;  // 48 MiB
  _Float16* Qh   = (_Float16*)(ws + off); off += (size_t)nX * 2;          // 16 MiB
  _Float16* Kh   = (_Float16*)(ws + off); off += (size_t)nX * 2;          // 16 MiB
  _Float16* Vh   = (_Float16*)(ws + off); off += (size_t)nX * 2;          // 16 MiB
  _Float16* Yh   = Xh;  // Xh dead after QKV GEMM

  cvt_f32_to_f16<<<(nX  + 255) / 256, 256, 0, stream>>>(x,      Xh,  nX);
  cvt_f32_to_f16<<<(nWa + 255) / 256, 256, 0, stream>>>(W_attn, Wah, nWa);
  cvt_f32_to_f16<<<(nWp + 255) / 256, 256, 0, stream>>>(W_proj, Wph, nWp);

  gemm_f16_kernel<_Float16><<<dim3(3 * CC / 64, M / 64), 128, 0, stream>>>(
      Xh, Wah, QKVh, M, 3 * CC, CC);

  {
    int n = BB * TT * NH * (HD / 2);
    rope_kernel<<<(n + 255) / 256, 256, 0, stream>>>(QKVh, Qh, Kh, Vh);
  }

  attn_kernel<<<dim3(TT / 64, NH, BB), 128, 0, stream>>>(Qh, Kh, Vh, Yh);

  gemm_f16_kernel<float><<<dim3(CC / 64, M / 64), 128, 0, stream>>>(
      Yh, Wph, out, M, CC, CC);
}
